// TixelNet_71648644432452
// MI455X (gfx1250) — matrix operand order repaired and checked by reading, not compiled, not run on hardware
//
#include <hip/hip_runtime.h>
#include <stdint.h>

// ---------------------------------------------------------------------------
// TixelNet: per-event positional-encoding MLP (10->32->32->16) + masked
// scatter-add into a (B, C, H, W) grid.  MLP runs on v_wmma_f32_16x16x32_f16.
//
// - Hidden layers N-split EVEN/ODD across the two 16x16 WMMA fragments so each
//   lane's two output columns (2*nlo, 2*nlo+1) are adjacent -> packed half2
//   ds_store_b32 (fused to ds_store_2addr_b32) repack between layers.
// - Each wave loops over 8 chunks of 32 events so the per-block setup
//   (weight staging + B/C fragment builds, held in VGPRs) amortizes 8x.
// ---------------------------------------------------------------------------

typedef __attribute__((ext_vector_type(16))) _Float16 v16h;
typedef __attribute__((ext_vector_type(8)))  _Float16 v8h;
typedef __attribute__((ext_vector_type(2)))  _Float16 v2h;
typedef __attribute__((ext_vector_type(8)))  float    v8f;

#define T_H   256
#define T_W   256
#define T_C   16
#define T_B   8
#define T_N   262144LL           // events per batch (2^18)
#define T_LOG2N 18
#define T_HW  (T_H * T_W)
#define WAVE_ITERS 8             // 32-event chunks per wave

// ---------------------------------------------------------------------------
// Fragment helpers (CDNA5 wave32 WMMA VGPR layouts, CDNA5 ISA)
// ---------------------------------------------------------------------------

// 16-bit A-matrix 16x32 (MxK): row M = lane%16.
// lanes 0-15 : halves 0-7 = K0-7,  halves 8-15 = K16-23
// lanes 16-31: halves 0-7 = K8-15, halves 8-15 = K24-31
__device__ __forceinline__ v16h load_A_frag(const _Float16* __restrict__ buf,
                                            int nlo, int hiHalf) {
  const _Float16* row = buf + nlo * 32;
  int k0 = hiHalf ? 8 : 0;
  v8h lo = *(const v8h*)(row + k0);        // ds_load_b128
  v8h hi = *(const v8h*)(row + k0 + 16);   // ds_load_b128
  return __builtin_shufflevector(lo, hi, 0, 1, 2, 3, 4, 5, 6, 7,
                                         8, 9, 10, 11, 12, 13, 14, 15);
}

// 16-bit B-matrix 32x16 (KxN): column N = `col` (caller supplies the map),
// lanes 0-15 hold K=0-7,16-23, lanes 16-31 hold K=8-15,24-31 (same as A).
__device__ __forceinline__ v16h load_B_frag(const _Float16* __restrict__ Wsh,
                                            int ncols, int col, int hiHalf) {
  v16h b;
  int kb = hiHalf ? 8 : 0;
#pragma unroll
  for (int j = 0; j < 16; ++j) b[j] = Wsh[(kb + j + (j >= 8 ? 8 : 0)) * ncols + col];
  return b;
}

// Bias broadcast as C operand: C(m,n) = bias[n] for every m (every VGPR).
__device__ __forceinline__ v8f bias_frag(float v) {
  v8f c;
#pragma unroll
  for (int r = 0; r < 8; ++r) c[r] = v;
  return c;
}

__device__ __forceinline__ v8f wmma_f16(v16h a, v16h b, v8f c) {
  return __builtin_amdgcn_wmma_f32_16x16x32_f16(false, a, false, b,
                                                (short)0, c, false, false);
}

// ---------------------------------------------------------------------------
// Output zeroing (d_out is poisoned by the harness before timing)
// ---------------------------------------------------------------------------
__global__ __launch_bounds__(256) void tixel_zero(float* __restrict__ out,
                                                  int n4) {
  int i = blockIdx.x * 256 + threadIdx.x;
  if (i < n4) ((float4*)out)[i] = make_float4(0.f, 0.f, 0.f, 0.f);
}

// ---------------------------------------------------------------------------
// Main kernel: 256 threads = 8 waves; each wave handles WAVE_ITERS*32 events.
// ---------------------------------------------------------------------------
__global__ __launch_bounds__(256) void tixel_mlp_scatter(
    const float*   __restrict__ times,
    const float*   __restrict__ pols,
    const float*   __restrict__ mask,
    const int64_t* __restrict__ pix,
    const float* __restrict__ W1, const float* __restrict__ b1,
    const float* __restrict__ W2, const float* __restrict__ b2,
    const float* __restrict__ W3, const float* __restrict__ b3,
    float* __restrict__ out) {
  // --- block-shared weight staging (f16) ---
  __shared__ _Float16 sW1[32 * 32];   // K-padded 10->32 with zeros
  __shared__ _Float16 sW2[32 * 32];
  __shared__ _Float16 sW3[32 * 16];
  __shared__ float    sB1[32], sB2[32], sB3[16];
  // --- per-wave activation tiles: 32 rows x 32 cols f16 (2 KB each) ---
  __shared__ _Float16 tilebuf[8][32 * 32];
  __shared__ float    smask[8][32];
  __shared__ int      spix[8][32];

  const int tid = threadIdx.x;
  for (int i = tid; i < 32 * 32; i += 256) {
    int k = i >> 5, n = i & 31;
    sW1[i] = (_Float16)((k < 10) ? W1[k * 32 + n] : 0.f);
    sW2[i] = (_Float16)W2[i];
  }
  for (int i = tid; i < 32 * 16; i += 256) sW3[i] = (_Float16)W3[i];
  if (tid < 32) { sB1[tid] = b1[tid]; sB2[tid] = b2[tid]; }
  if (tid < 16) sB3[tid] = b3[tid];
  __syncthreads();

  const int wave   = tid >> 5;
  const int lane   = tid & 31;
  const int nlo    = lane & 15;
  const int hiHalf = lane >> 4;   // 0 or 1
  const int moff   = hiHalf ? 8 : 0;

  // --- per-wave constant fragments (held in VGPRs across the whole kernel) --
  // Hidden layers: fragment a = even columns (2*nlo), b = odd (2*nlo+1).
  const int ce = 2 * nlo, co = 2 * nlo + 1;
  const v16h B1a = load_B_frag(sW1, 32, ce, hiHalf);
  const v16h B1b = load_B_frag(sW1, 32, co, hiHalf);
  const v16h B2a = load_B_frag(sW2, 32, ce, hiHalf);
  const v16h B2b = load_B_frag(sW2, 32, co, hiHalf);
  const v16h B3  = load_B_frag(sW3, 16, nlo, hiHalf);
  const v8f  C1a = bias_frag(sB1[ce]);
  const v8f  C1b = bias_frag(sB1[co]);
  const v8f  C2a = bias_frag(sB2[ce]);
  const v8f  C2b = bias_frag(sB2[co]);
  const v8f  C3  = bias_frag(sB3[nlo]);

  const long long wavebase =
      ((long long)blockIdx.x * 8 + wave) * (32LL * WAVE_ITERS);

#pragma unroll 1
  for (int it = 0; it < WAVE_ITERS; ++it) {
    const long long base = wavebase + (long long)it * 32;

    // --- stage 1: each lane computes one event's feature row ---------------
    {
      long long e = base + lane;
      float t = times[e];
      float p = pols[e];
      smask[wave][lane] = mask[e];
      spix[wave][lane]  = (int)pix[e];

      float c0 = __cosf(t * 1.f),   c1 = __cosf(t * 2.f);
      float c2 = __cosf(t * 4.f),   c3 = __cosf(t * 8.f);
      float c4 = __cosf(t * 16.f),  c5 = __cosf(t * 32.f);
      float c6 = __cosf(t * 64.f),  c7 = __cosf(t * 128.f);

      v8h f0 = { (_Float16)t,  (_Float16)c0, (_Float16)c1, (_Float16)c2,
                 (_Float16)c3, (_Float16)c4, (_Float16)c5, (_Float16)c6 };
      v8h f1 = { (_Float16)c7, (_Float16)p, (_Float16)0.f, (_Float16)0.f,
                 (_Float16)0.f, (_Float16)0.f, (_Float16)0.f, (_Float16)0.f };
      v8h fz = { (_Float16)0.f, (_Float16)0.f, (_Float16)0.f, (_Float16)0.f,
                 (_Float16)0.f, (_Float16)0.f, (_Float16)0.f, (_Float16)0.f };

      _Float16* row = &tilebuf[wave][lane * 32];
      *(v8h*)(row + 0)  = f0;   // ds_store_b128 x4
      *(v8h*)(row + 8)  = f1;
      *(v8h*)(row + 16) = fz;
      *(v8h*)(row + 24) = fz;
    }
    // all 32 lanes of this wave wrote; LDS is in-order per wave, make the
    // compiler/HW honor store->load ordering before the cross-lane A loads.
    asm volatile("s_wait_dscnt 0x0" ::: "memory");

    // --- stage 2: two 16-event WMMA tiles per wave -------------------------
#pragma unroll
    for (int t = 0; t < 2; ++t) {
      _Float16* buf = &tilebuf[wave][t * 16 * 32];

      // layer 1: feats(16x32) x W1(32x32) + b1, ReLU
      v16h A  = load_A_frag(buf, nlo, hiHalf);
      v8f  D0 = wmma_f16(A, B1a, C1a);   // even columns
      v8f  D1 = wmma_f16(A, B1b, C1b);   // odd  columns
#pragma unroll
      for (int r = 0; r < 8; ++r) {
        v2h pk = { (_Float16)fmaxf(D0[r], 0.f), (_Float16)fmaxf(D1[r], 0.f) };
        *(v2h*)&buf[(r + moff) * 32 + ce] = pk;   // packed cols 2n,2n+1
      }
      asm volatile("s_wait_dscnt 0x0" ::: "memory");

      // layer 2: h1(16x32) x W2(32x32) + b2, ReLU
      A  = load_A_frag(buf, nlo, hiHalf);
      D0 = wmma_f16(A, B2a, C2a);
      D1 = wmma_f16(A, B2b, C2b);
#pragma unroll
      for (int r = 0; r < 8; ++r) {
        v2h pk = { (_Float16)fmaxf(D0[r], 0.f), (_Float16)fmaxf(D1[r], 0.f) };
        *(v2h*)&buf[(r + moff) * 32 + ce] = pk;
      }
      asm volatile("s_wait_dscnt 0x0" ::: "memory");

      // layer 3: h2(16x32) x W3(32x16) + b3 -> (16 events x 16 channels) f32
      A = load_A_frag(buf, nlo, hiHalf);
      v8f D = wmma_f16(A, B3, C3);

      // scatter: lane holds channel n=nlo for events m = moff+r of the tile.
      // out is (B, C, H*W); event batch = e >> 18.
#pragma unroll
      for (int r = 0; r < 8; ++r) {
        int m32     = t * 16 + moff + r;         // event index within chunk
        long long e = base + m32;
        int bidx    = (int)(e >> T_LOG2N);
        float ev    = D[r] * smask[wave][m32];
        int p       = spix[wave][m32];
        atomicAdd(out + ((size_t)(bidx * T_C + nlo)) * T_HW + p, ev);
      }
    }
  }
}

// ---------------------------------------------------------------------------
// Launcher
// ---------------------------------------------------------------------------
extern "C" void kernel_launch(void* const* d_in, const int* in_sizes, int n_in,
                              void* d_out, int out_size, void* d_ws,
                              size_t ws_size, hipStream_t stream) {
  const float*   times = (const float*)d_in[0];
  const float*   pols  = (const float*)d_in[1];
  const float*   msk   = (const float*)d_in[2];
  const int64_t* pix   = (const int64_t*)d_in[3];
  const float*   W1    = (const float*)d_in[4];
  const float*   b1    = (const float*)d_in[5];
  const float*   W2    = (const float*)d_in[6];
  const float*   b2    = (const float*)d_in[7];
  const float*   W3    = (const float*)d_in[8];
  const float*   b3    = (const float*)d_in[9];
  float*         out   = (float*)d_out;

  // zero the (B,C,H,W) accumulation grid (harness poisons d_out)
  int n4      = out_size / 4;                 // out_size = 8*16*256*256
  int zblocks = (n4 + 255) / 256;
  tixel_zero<<<zblocks, 256, 0, stream>>>(out, n4);

  // 8 waves/block, WAVE_ITERS*32 events per wave -> 2048 events per block
  long long total_events = T_B * T_N;                       // 2,097,152
  int blocks = (int)(total_events / (256LL * WAVE_ITERS));  // 1024
  tixel_mlp_scatter<<<blocks, 256, 0, stream>>>(times, pols, msk, pix,
                                                W1, b1, W2, b2, W3, b3, out);
}
